// E2ECR_18700287607062
// MI455X (gfx1250) — matrix operand order repaired and checked by reading, not compiled, not run on hardware
//
#include <hip/hip_runtime.h>

typedef __attribute__((ext_vector_type(16))) __bf16   v16bf;
typedef __attribute__((ext_vector_type(8)))  float    v8f;
typedef __attribute__((ext_vector_type(8)))  unsigned v8u;

#define T_    2048
#define E_    768
#define S_    256
#define H_    150
#define HP_   160     // H padded to multiple of 16
#define NT_   10      // HP_/16 : h tiles / waves per block
#define E3_   2304
#define KT_E  24      // 768/32
#define KT_G  72      // 2304/32
#define KT_H  5       // 160/32
#define JT_   4       // j-tiles per k4 block (M-block = 64 rows)
#define KC_   18      // K-tiles per staged chunk (72/4)

// A-fragment K-order permutation within a 32-block (involution):
// positions [0-7,8-15,16-23,24-31] -> k offsets [0-7,16-23,8-15,24-31]
__device__ __forceinline__ int srcoff(int u) {
    int g = (u >> 3) & 3;
    int base = (g == 1) ? 16 : (g == 2) ? 8 : (g == 3) ? 24 : 0;
    return base + (u & 7);
}

__device__ __forceinline__ __bf16 f2bf(float f) {
    union { float f; unsigned u; } x; x.f = f;
    unsigned r = x.u + 0x7FFFu + ((x.u >> 16) & 1u);   // round-to-nearest-even
    unsigned short h = (unsigned short)(r >> 16);
    union { unsigned short s; __bf16 b; } y; y.s = h;
    return y.b;
}

__device__ __forceinline__ v8f wmma_bf16(v16bf a, v16bf b, v8f c) {
    return __builtin_amdgcn_wmma_f32_16x16x32_bf16(false, a, false, b, (short)0, c, false, false);
}

// packed bf16 multiply (VOP3P v_pk_mul_bf16) on 8 dwords = 16 bf16 lanes
__device__ __forceinline__ v8u pk_mul_bf16x16(v8u a, v8u b) {
    v8u r;
#pragma unroll
    for (int q = 0; q < 8; ++q) {
        unsigned d;
        asm("v_pk_mul_bf16 %0, %1, %2" : "=v"(d) : "v"(a[q]), "v"(b[q]));
        r[q] = d;
    }
    return r;
}

// ---------------------------------------------------------------------------
// Pack a KxN f32 weight (row-major) into bf16 B-fragment layout.
// ---------------------------------------------------------------------------
__global__ void pack_b_kernel(const float* __restrict__ W, __bf16* __restrict__ dst,
                              int K, int N, int KT, int NT) {
    int idx = blockIdx.x * 256 + threadIdx.x;
    int total = KT * NT * 512;
    if (idx >= total) return;
    int tt   = idx & 15;
    int lane = (idx >> 4) & 31;
    int ktnt = idx >> 9;
    int nt = ktnt % NT;
    int kt = ktnt / NT;
    int k = kt * 32 + srcoff(((lane >> 4) << 4) + tt);
    int n = nt * 16 + (lane & 15);
    float v = (k < K && n < N) ? W[(size_t)k * N + n] : 0.0f;
    dst[idx] = f2bf(v);
}

// ---------------------------------------------------------------------------
// Pack encoded_docs (2048x768) into bf16 A-fragment layout, 16-row tiles.
// ---------------------------------------------------------------------------
__global__ void pack_eA_kernel(const float* __restrict__ e, __bf16* __restrict__ dst) {
    int idx = blockIdx.x * 256 + threadIdx.x;
    if (idx >= 128 * KT_E * 512) return;
    int tt   = idx & 15;
    int lane = (idx >> 4) & 31;
    int rk   = idx >> 9;
    int kt = rk % KT_E;
    int rt = rk / KT_E;
    int m   = lane & 15;
    int hiL = lane >> 4;
    int k = kt * 32 + srcoff(hiL * 16 + tt);
    dst[idx] = f2bf(e[(size_t)(rt * 16 + m) * E_ + k]);
}

// ---------------------------------------------------------------------------
// Kernel 1: attention scores. 128 blocks x 320 threads.
// ---------------------------------------------------------------------------
__global__ __launch_bounds__(320) void k1_attn(
    const __bf16* __restrict__ eA, const __bf16* __restrict__ aW1p,
    const __bf16* __restrict__ aW2p, const float* __restrict__ ab1,
    const float* __restrict__ ab2, const float* __restrict__ aW3,
    const float* __restrict__ ab3, float* __restrict__ attn) {
    __shared__ __align__(32) __bf16 hA[KT_H * 32 * 16];
    __shared__ float h2l[16 * HP_];
    int lane = threadIdx.x & 31;
    int w    = threadIdx.x >> 5;
    int blk  = blockIdx.x;
    int hiL  = lane >> 4;
    int m0   = hiL << 3;
    int n    = w * 16 + (lane & 15);

    v8f c = {};
    for (int kt = 0; kt < KT_E; ++kt) {
        v16bf a = *(const v16bf*)(eA + ((size_t)(blk * KT_E + kt) * 32 + lane) * 16);
        v16bf b = *(const v16bf*)(aW1p + ((size_t)(kt * NT_ + w) * 32 + lane) * 16);
        c = wmma_bf16(a, b, c);
    }
    float bias1 = (n < H_) ? ab1[n] : 0.0f;
    int kt2 = n >> 5;
    int pos = srcoff(n & 31);
#pragma unroll
    for (int v = 0; v < 8; ++v) {
        int m = v + m0;
        float h = c[v] + bias1;
        h = h > 0.0f ? h : 0.0f;
        hA[((kt2 * 32 + ((pos >> 4) << 4) + m) << 4) + (pos & 15)] = f2bf(h);
    }
    __syncthreads();

    v8f c2 = {};
    for (int kt = 0; kt < KT_H; ++kt) {
        v16bf a = *(const v16bf*)&hA[(kt * 32 + lane) * 16];
        v16bf b = *(const v16bf*)(aW2p + ((size_t)(kt * NT_ + w) * 32 + lane) * 16);
        c2 = wmma_bf16(a, b, c2);
    }
    float bias2 = (n < H_) ? ab2[n] : 0.0f;
#pragma unroll
    for (int v = 0; v < 8; ++v) {
        float h = c2[v] + bias2;
        h2l[(v + m0) * HP_ + n] = h > 0.0f ? h : 0.0f;
    }
    __syncthreads();

    if (threadIdx.x < 16) {
        int m = threadIdx.x;
        float s = ab3[0];
        for (int h = 0; h < H_; ++h) s += h2l[m * HP_ + h] * aW3[h];
        attn[blk * 16 + m] = s;
    }
}

// ---------------------------------------------------------------------------
// Kernel 2: build g and emit bf16 copies: gPb (row-major, srcoff-permuted) and
// gAb (A-fragment layout).
// ---------------------------------------------------------------------------
__global__ __launch_bounds__(256) void k2_g(
    const float* __restrict__ e, const float* __restrict__ attn,
    const int* __restrict__ sstart, const int* __restrict__ swidth,
    __bf16* __restrict__ gPb, __bf16* __restrict__ gAb) {
    __shared__ float gl[E3_];
    int s  = blockIdx.x;
    int st = sstart[s];
    int en = st + swidth[s];
    if (en > T_ - 1) en = T_ - 1;

    for (int cc = threadIdx.x; cc < E_; cc += 256) {
        gl[cc]        = e[(size_t)st * E_ + cc];
        gl[E_ + cc]   = e[(size_t)en * E_ + cc];
        float acc = 0.0f;
        for (int t = st; t <= en; ++t) acc += e[(size_t)t * E_ + cc] * attn[t];
        gl[2 * E_ + cc] = acc;
    }
    __syncthreads();

    int rt = s >> 4, m = s & 15;
    for (int u = threadIdx.x; u < E3_; u += 256) {
        int kt  = u >> 5;
        int pos = u & 31;
        __bf16 v = f2bf(gl[kt * 32 + srcoff(pos)]);
        gPb[(size_t)s * E3_ + u] = v;
        gAb[((size_t)(rt * KT_G + kt) * 32 + ((pos >> 4) << 4) + m) * 16 + (pos & 15)] = v;
    }
}

// ---------------------------------------------------------------------------
// Kernel 3: hi = g@Wa, hj = g@Wb, m = MLP_m(g). 16 blocks x 320 threads.
// ---------------------------------------------------------------------------
__global__ __launch_bounds__(320) void k3_span(
    const __bf16* __restrict__ gAb, const __bf16* __restrict__ Wap,
    const __bf16* __restrict__ Wbp, const __bf16* __restrict__ mW1p,
    const __bf16* __restrict__ mW2p, const float* __restrict__ mb1,
    const float* __restrict__ mb2, const float* __restrict__ mW3,
    const float* __restrict__ mb3, float* __restrict__ hi_o,
    float* __restrict__ hj_o, float* __restrict__ mval) {
    __shared__ __align__(32) __bf16 hA[KT_H * 32 * 16];
    __shared__ float h2l[16 * HP_];
    int lane = threadIdx.x & 31;
    int w    = threadIdx.x >> 5;
    int blk  = blockIdx.x;
    int hiL  = lane >> 4;
    int m0   = hiL << 3;
    int n    = w * 16 + (lane & 15);

    v8f ci = {}, cj = {}, cm = {};
    for (int kt = 0; kt < KT_G; ++kt) {
        v16bf a  = *(const v16bf*)(gAb + ((size_t)(blk * KT_G + kt) * 32 + lane) * 16);
        v16bf ba = *(const v16bf*)(Wap  + ((size_t)(kt * NT_ + w) * 32 + lane) * 16);
        v16bf bb = *(const v16bf*)(Wbp  + ((size_t)(kt * NT_ + w) * 32 + lane) * 16);
        v16bf bm = *(const v16bf*)(mW1p + ((size_t)(kt * NT_ + w) * 32 + lane) * 16);
        ci = wmma_bf16(a, ba, ci);
        cj = wmma_bf16(a, bb, cj);
        cm = wmma_bf16(a, bm, cm);
    }
#pragma unroll
    for (int v = 0; v < 8; ++v) {
        int row = blk * 16 + v + m0;
        hi_o[(size_t)row * HP_ + n] = ci[v];
        hj_o[(size_t)row * HP_ + n] = cj[v];
    }
    float bias1 = (n < H_) ? mb1[n] : 0.0f;
    int kt2 = n >> 5;
    int pos = srcoff(n & 31);
#pragma unroll
    for (int v = 0; v < 8; ++v) {
        int m = v + m0;
        float h = cm[v] + bias1;
        h = h > 0.0f ? h : 0.0f;
        hA[((kt2 * 32 + ((pos >> 4) << 4) + m) << 4) + (pos & 15)] = f2bf(h);
    }
    __syncthreads();

    v8f c2 = {};
    for (int kt = 0; kt < KT_H; ++kt) {
        v16bf a = *(const v16bf*)&hA[(kt * 32 + lane) * 16];
        v16bf b = *(const v16bf*)(mW2p + ((size_t)(kt * NT_ + w) * 32 + lane) * 16);
        c2 = wmma_bf16(a, b, c2);
    }
    float bias2 = (n < H_) ? mb2[n] : 0.0f;
#pragma unroll
    for (int v = 0; v < 8; ++v) {
        float h = c2[v] + bias2;
        h2l[(v + m0) * HP_ + n] = h > 0.0f ? h : 0.0f;
    }
    __syncthreads();

    if (threadIdx.x < 16) {
        int m = threadIdx.x;
        float s = mb3[0];
        for (int h = 0; h < H_; ++h) s += h2l[m * HP_ + h] * mW3[h];
        mval[blk * 16 + m] = s;
    }
}

// ---------------------------------------------------------------------------
// Kernel 4: pair kernel, M-block = 64 rows. blockIdx = i*4 + jc, 320 threads.
// Per K-chunk: stage P = g[j,:]*g[i,:] (bf16, A-frag layout) into LDS via
// v_pk_mul_bf16, then each wave streams one B frag/kt and fires 4 WMMAs.
// ---------------------------------------------------------------------------
__global__ __launch_bounds__(320) void k4_pair(
    const __bf16* __restrict__ gAb, const __bf16* __restrict__ gPb,
    const __bf16* __restrict__ Wcp, const __bf16* __restrict__ pW2p,
    const float* __restrict__ hi_i, const float* __restrict__ hj_i,
    const float* __restrict__ pb1, const float* __restrict__ pb2,
    const float* __restrict__ pW3, const float* __restrict__ pb3,
    const float* __restrict__ mval, float* __restrict__ out) {
    __shared__ __align__(32) __bf16 ldsP[JT_ * KC_ * 32 * 16];   // 72 KB
    __shared__ __align__(32) __bf16 hA[KT_H * 32 * 16];          // 5 KB
    __shared__ float h2l[16 * HP_];                              // 10 KB
    int i  = blockIdx.x >> 2;
    int jc = blockIdx.x & 3;
    int lane = threadIdx.x & 31;
    int w    = threadIdx.x >> 5;
    int hiL  = lane >> 4;
    int m0   = hiL << 3;
    int n    = w * 16 + (lane & 15);
    const __bf16* gPi = gPb + (size_t)i * E3_;

    v8f c[JT_] = {};
    for (int kc = 0; kc < KT_G / KC_; ++kc) {
        // stage P fragments for this K-chunk (frags split across the 10 waves)
        for (int f = w; f < JT_ * KC_; f += NT_) {
            int jt  = f / KC_;
            int kt0 = f % KC_;
            int kt  = kc * KC_ + kt0;
            int jtg = jc * JT_ + jt;
            v8u a = *(const v8u*)(gAb + ((size_t)(jtg * KT_G + kt) * 32 + lane) * 16);
            v8u s = *(const v8u*)(gPi + kt * 32 + (hiL << 4));
            v8u p = pk_mul_bf16x16(a, s);
            *(v8u*)&ldsP[((jt * KC_ + kt0) * 32 + lane) * 16] = p;
        }
        __syncthreads();
        // consume: one B fragment per kt, 4 WMMAs against LDS-resident A tiles
        for (int kt0 = 0; kt0 < KC_; ++kt0) {
            int kt = kc * KC_ + kt0;
            v16bf b = *(const v16bf*)(Wcp + ((size_t)(kt * NT_ + w) * 32 + lane) * 16);
#pragma unroll
            for (int jt = 0; jt < JT_; ++jt) {
                v16bf a = *(const v16bf*)&ldsP[((jt * KC_ + kt0) * 32 + lane) * 16];
                c[jt] = wmma_bf16(a, b, c[jt]);
            }
        }
        __syncthreads();
    }

    float hival = hi_i[(size_t)i * HP_ + n];
    float bias1 = (n < H_) ? pb1[n] : 0.0f;
    float bias2 = (n < H_) ? pb2[n] : 0.0f;
    int kt2 = n >> 5;
    int pos = srcoff(n & 31);

    for (int jt = 0; jt < JT_; ++jt) {
        int jtg = jc * JT_ + jt;
#pragma unroll
        for (int v = 0; v < 8; ++v) {
            int m = v + m0;
            float h = c[jt][v] + hival + hj_i[(size_t)(jtg * 16 + m) * HP_ + n] + bias1;
            h = h > 0.0f ? h : 0.0f;
            hA[((kt2 * 32 + ((pos >> 4) << 4) + m) << 4) + (pos & 15)] = f2bf(h);
        }
        __syncthreads();

        v8f c2 = {};
        for (int kt = 0; kt < KT_H; ++kt) {
            v16bf a = *(const v16bf*)&hA[(kt * 32 + lane) * 16];
            v16bf b = *(const v16bf*)(pW2p + ((size_t)(kt * NT_ + w) * 32 + lane) * 16);
            c2 = wmma_bf16(a, b, c2);
        }
#pragma unroll
        for (int v = 0; v < 8; ++v) {
            float h = c2[v] + bias2;
            h2l[(v + m0) * HP_ + n] = h > 0.0f ? h : 0.0f;
        }
        __syncthreads();

        if (threadIdx.x < 16) {
            int m = threadIdx.x;
            int j = jtg * 16 + m;
            float s = pb3[0];
            for (int h = 0; h < H_; ++h) s += h2l[m * HP_ + h] * pW3[h];
            float ext = (mval[i] + mval[j] + s) * (1.0f / 3.0f);
            ext = ext < 0.0f ? 0.0f : (ext > 1.0f ? 1.0f : ext);
            out[(size_t)i * S_ + j] = ext;
        }
        __syncthreads();
    }
}

// ---------------------------------------------------------------------------
extern "C" void kernel_launch(void* const* d_in, const int* in_sizes, int n_in,
                              void* d_out, int out_size, void* d_ws, size_t ws_size,
                              hipStream_t stream) {
    (void)in_sizes; (void)n_in; (void)out_size; (void)ws_size;
    const float* e   = (const float*)d_in[0];
    const int*   sst = (const int*)d_in[1];
    const int*   swd = (const int*)d_in[2];
    const float* aW1 = (const float*)d_in[3];
    const float* ab1 = (const float*)d_in[4];
    const float* aW2 = (const float*)d_in[5];
    const float* ab2 = (const float*)d_in[6];
    const float* aW3 = (const float*)d_in[7];
    const float* ab3 = (const float*)d_in[8];
    const float* mW1 = (const float*)d_in[9];
    const float* mb1 = (const float*)d_in[10];
    const float* mW2 = (const float*)d_in[11];
    const float* mb2 = (const float*)d_in[12];
    const float* mW3 = (const float*)d_in[13];
    const float* mb3 = (const float*)d_in[14];
    const float* pW1 = (const float*)d_in[15];
    const float* pb1 = (const float*)d_in[16];
    const float* pW2 = (const float*)d_in[17];
    const float* pb2 = (const float*)d_in[18];
    const float* pW3 = (const float*)d_in[19];
    const float* pb3 = (const float*)d_in[20];
    float* out = (float*)d_out;

    char* ws = (char*)d_ws;
    size_t off = 0;
    auto take = [&](size_t bytes) {
        size_t o = off;
        off += (bytes + 255) & ~(size_t)255;
        return o;
    };
    float*  attn = (float*)(ws + take((size_t)T_ * 4));
    float*  mval = (float*)(ws + take((size_t)S_ * 4));
    float*  hi_b = (float*)(ws + take((size_t)S_ * HP_ * 4));
    float*  hj_b = (float*)(ws + take((size_t)S_ * HP_ * 4));
    __bf16* gPb  = (__bf16*)(ws + take((size_t)S_ * E3_ * 2));
    __bf16* gAb  = (__bf16*)(ws + take((size_t)16 * KT_G * 512 * 2));
    __bf16* eA   = (__bf16*)(ws + take((size_t)128 * KT_E * 512 * 2));
    __bf16* aW1p = (__bf16*)(ws + take((size_t)KT_E * NT_ * 512 * 2));
    __bf16* aW2p = (__bf16*)(ws + take((size_t)KT_H * NT_ * 512 * 2));
    __bf16* mW1p = (__bf16*)(ws + take((size_t)KT_G * NT_ * 512 * 2));
    __bf16* mW2p = (__bf16*)(ws + take((size_t)KT_H * NT_ * 512 * 2));
    __bf16* Wap  = (__bf16*)(ws + take((size_t)KT_G * NT_ * 512 * 2));
    __bf16* Wbp  = (__bf16*)(ws + take((size_t)KT_G * NT_ * 512 * 2));
    __bf16* Wcp  = (__bf16*)(ws + take((size_t)KT_G * NT_ * 512 * 2));
    __bf16* pW2p = (__bf16*)(ws + take((size_t)KT_H * NT_ * 512 * 2));

    auto packB = [&](const float* W, __bf16* dst, int K, int N, int KT, int NT) {
        int total = KT * NT * 512;
        pack_b_kernel<<<(total + 255) / 256, 256, 0, stream>>>(W, dst, K, N, KT, NT);
    };
    packB(aW1, aW1p, E_,  H_, KT_E, NT_);
    packB(aW2, aW2p, H_,  H_, KT_H, NT_);
    packB(mW1, mW1p, E3_, H_, KT_G, NT_);
    packB(mW2, mW2p, H_,  H_, KT_H, NT_);
    packB(pW1,                        Wap, E3_, H_, KT_G, NT_);
    packB(pW1 + (size_t)E3_ * H_,     Wbp, E3_, H_, KT_G, NT_);
    packB(pW1 + (size_t)2 * E3_ * H_, Wcp, E3_, H_, KT_G, NT_);
    packB(pW2, pW2p, H_, H_, KT_H, NT_);

    pack_eA_kernel<<<(128 * KT_E * 512 + 255) / 256, 256, 0, stream>>>(e, eA);

    k1_attn<<<128, 320, 0, stream>>>(eA, aW1p, aW2p, ab1, ab2, aW3, ab3, attn);
    k2_g<<<256, 256, 0, stream>>>(e, attn, sst, swd, gPb, gAb);
    k3_span<<<16, 320, 0, stream>>>(gAb, Wap, Wbp, mW1p, mW2p, mb1, mb2, mW3, mb3,
                                    hi_b, hj_b, mval);
    k4_pair<<<1024, 320, 0, stream>>>(gAb, gPb, Wcp, pW2p, hi_b, hj_b, pb1, pb2,
                                      pW3, pb3, mval, out);
}
